// MLPWithSinkhorn_81046032875888
// MI455X (gfx1250) — matrix operand order repaired and checked by reading, not compiled, not run on hardware
//
#include <hip/hip_runtime.h>

// ============================================================
// MLP + Sinkhorn for MI455X (gfx1250), wave32, WMMA f16->f32.
// GEMMs: global -> LDS via global_load_async_to_lds_b128 (ASYNCcnt),
// WMMA fragments read from LDS, v_wmma_f32_16x16x32_f16 compute.
// Sinkhorn/BN/exp in fp32. K matrix (64MB) is L2-resident (192MB L2).
// ============================================================

typedef _Float16 h16;
typedef __attribute__((ext_vector_type(16))) _Float16 v16h;
typedef __attribute__((ext_vector_type(8)))  _Float16 v8h;
typedef __attribute__((ext_vector_type(8)))  float    v8f;

#define BN_EPS    1e-5f
#define SK_EPS    0.1f
#define SK_ITERS  50

#define N_BATCH   4096
#define D_IN      1280
#define H1_DIM    2048
#define H2_DIM    1024
#define H3_DIM    512
#define D_OUT     512

// issue one per-lane async global->LDS 16B copy (tracked by ASYNCcnt)
__device__ __forceinline__ void async_cp16(unsigned ldsOff, const h16* g) {
  asm volatile("global_load_async_to_lds_b128 %0, %1, off"
               :: "v"(ldsOff), "v"((unsigned long long)(uintptr_t)g)
               : "memory");
}
__device__ __forceinline__ void wait_async0() {
  asm volatile("s_wait_asynccnt 0x0" ::: "memory");
}

// ---------------------------------------------------------------
// WMMA NT GEMM:  C[M,N] = A[M,K] * B[N,K]^T   (torch x @ W.T form)
// A,B: f16 row-major. C: fp32 row-major.
// Epilogues: 0 = none, 1 = +bias[col], 2 = exp(-(sq[r]+sq[c]-2acc)/eps)
// Block: 256 thr = 8 waves; wave = 2x2 tiles of 16x16 (32x32);
// block tile 128(M) x 64(N), K-step 32, tiles staged in LDS.
// Requires M%128==0, N%64==0, K%32==0.
// ---------------------------------------------------------------
enum { EPI_NONE = 0, EPI_BIAS = 1, EPI_EXPCOST = 2 };

#define LDS_STRIDE 40            // 32 halfs + 8 pad -> 80B rows, 16B aligned
#define LDS_BOFF   (128 * LDS_STRIDE)

template <int EPI>
__global__ __launch_bounds__(256)
void gemm_nt_wmma(const h16* __restrict__ A, const h16* __restrict__ B,
                  float* __restrict__ C, int M, int N, int K,
                  const float* __restrict__ bias,
                  const float* __restrict__ sq)
{
  __shared__ __align__(16) h16 smem[LDS_BOFF + 64 * LDS_STRIDE]; // 15.4 KB

  const int t    = threadIdx.x;
  const int lane = t & 31;
  const int wave = t >> 5;
  const int half = lane >> 4;      // which 16-lane half
  const int r    = lane & 15;      // row (A) / col (B,C) within tile
  const int wm   = wave >> 1;      // 0..3
  const int wn   = wave & 1;       // 0..1
  const int m0b  = blockIdx.y * 128;
  const int n0b  = blockIdx.x * 64;

  // LDS byte offset of smem base (generic LDS addr low 32 bits = LDS offset)
  const unsigned smemBase = (unsigned)(uintptr_t)&smem[0];

  v8f acc[2][2];
#pragma unroll
  for (int mi = 0; mi < 2; ++mi)
#pragma unroll
    for (int ni = 0; ni < 2; ++ni)
#pragma unroll
      for (int j = 0; j < 8; ++j) acc[mi][ni][j] = 0.0f;

  for (int kk = 0; kk < K; kk += 32) {
    __syncthreads();   // previous tile fully consumed before overwrite

    // ---- stage A tile 128x32 (512 x 16B chunks, 2 per thread) ----
#pragma unroll
    for (int q = 0; q < 2; ++q) {
      int c = t + q * 256;
      int row = c >> 2, seg = c & 3;
      const h16* g = A + (size_t)(m0b + row) * K + kk + seg * 8;
      async_cp16(smemBase + (unsigned)((row * LDS_STRIDE + seg * 8) * 2), g);
    }
    // ---- stage B tile 64x32 (256 x 16B chunks, 1 per thread) ----
    {
      int row = t >> 2, seg = t & 3;
      const h16* g = B + (size_t)(n0b + row) * K + kk + seg * 8;
      async_cp16(smemBase + (unsigned)((LDS_BOFF + row * LDS_STRIDE + seg * 8) * 2), g);
    }
    wait_async0();
    __syncthreads();

    // ---- fragments from LDS ----
    v16h af[2], bf[2];
    // A frag: lane(half,r): K = 8*half + j (j<8), 16 + 8*half + (j-8)
#pragma unroll
    for (int mi = 0; mi < 2; ++mi) {
      const int arow = wm * 32 + mi * 16 + r;
      const v8h* pa = (const v8h*)&smem[arow * LDS_STRIDE + half * 8];
      v8h lo = pa[0];
      v8h hi = *(const v8h*)&smem[arow * LDS_STRIDE + 16 + half * 8];
#pragma unroll
      for (int j = 0; j < 8; ++j) { af[mi][j] = lo[j]; af[mi][8 + j] = hi[j]; }
    }
    // B frag: lane column = r, K = 8*half + j (j<8), 16 + 8*half + (j-8)
#pragma unroll
    for (int ni = 0; ni < 2; ++ni) {
      const int brow = wn * 32 + ni * 16 + r;
      v8h blo = *(const v8h*)&smem[LDS_BOFF + brow * LDS_STRIDE + half * 8];
      v8h bhi = *(const v8h*)&smem[LDS_BOFF + brow * LDS_STRIDE + 16 + half * 8];
#pragma unroll
      for (int j = 0; j < 8; ++j) { bf[ni][j] = blo[j]; bf[ni][8 + j] = bhi[j]; }
    }

#pragma unroll
    for (int mi = 0; mi < 2; ++mi)
#pragma unroll
      for (int ni = 0; ni < 2; ++ni)
        acc[mi][ni] = __builtin_amdgcn_wmma_f32_16x16x32_f16(
            false, af[mi], false, bf[ni], (short)0, acc[mi][ni], false, false);
  }

  // Epilogue + store. C/D layout: VGPR v -> row = v + 8*half, col = r.
#pragma unroll
  for (int mi = 0; mi < 2; ++mi) {
#pragma unroll
    for (int ni = 0; ni < 2; ++ni) {
      const int col = n0b + wn * 32 + ni * 16 + r;
      float addc = 0.0f, sqc = 0.0f;
      if (EPI == EPI_BIAS)    addc = bias[col];
      if (EPI == EPI_EXPCOST) sqc  = sq[col];
      const int rowbase = m0b + wm * 32 + mi * 16 + half * 8;
#pragma unroll
      for (int v = 0; v < 8; ++v) {
        const int row = rowbase + v;
        float cv = acc[mi][ni][v];
        if (EPI == EPI_BIAS)    cv += addc;
        if (EPI == EPI_EXPCOST) {
          float cost = sq[row] + sqc - 2.0f * cv;     // squared distance
          cv = expf(-cost * (1.0f / SK_EPS));         // K = exp(-C/eps)
        }
        C[(size_t)row * N + col] = cv;
      }
    }
  }
}

// ---------------------------------------------------------------
// Elementwise / reduction helpers (all fp32, deterministic)
// ---------------------------------------------------------------
__global__ void f32_to_f16_k(const float* __restrict__ in, h16* __restrict__ out, size_t n) {
  size_t i = (size_t)blockIdx.x * blockDim.x + threadIdx.x;
  size_t s = (size_t)gridDim.x * blockDim.x;
  for (; i < n; i += s) out[i] = (h16)in[i];
}

// BN stage 1: partial column sums over a row slab. grid=(H/256, SLABS)
#define BN_SLABS 16
__global__ void bn_partial_k(const float* __restrict__ Y, float* __restrict__ psum,
                             float* __restrict__ psq, int M, int H) {
  const int col = blockIdx.x * blockDim.x + threadIdx.x;
  const int rows = M / BN_SLABS;
  const int r0 = blockIdx.y * rows;
  float s = 0.0f, s2 = 0.0f;
  for (int rr = r0; rr < r0 + rows; ++rr) {
    float x = Y[(size_t)rr * H + col];
    s += x; s2 += x * x;
  }
  psum[(size_t)blockIdx.y * H + col] = s;
  psq [(size_t)blockIdx.y * H + col] = s2;
}

// BN stage 2: reduce slabs -> sum, sumsq (deterministic order)
__global__ void bn_reduce_k(const float* __restrict__ psum, const float* __restrict__ psq,
                            float* __restrict__ sum, float* __restrict__ sumsq, int H) {
  const int col = blockIdx.x * blockDim.x + threadIdx.x;
  if (col >= H) return;
  float s = 0.0f, s2 = 0.0f;
  for (int b = 0; b < BN_SLABS; ++b) {
    s  += psum[(size_t)b * H + col];
    s2 += psq [(size_t)b * H + col];
  }
  sum[col] = s; sumsq[col] = s2;
}

// BN apply + ReLU; writes fp32 (optional) and f16. H must be pow2 (hmask=H-1).
__global__ void bn_relu_k(const float* __restrict__ Y,
                          const float* __restrict__ sum, const float* __restrict__ sumsq,
                          const float* __restrict__ gamma, const float* __restrict__ beta,
                          float* __restrict__ outF, h16* __restrict__ outH,
                          size_t total, int hmask, float invM) {
  size_t i = (size_t)blockIdx.x * blockDim.x + threadIdx.x;
  size_t s = (size_t)gridDim.x * blockDim.x;
  for (; i < total; i += s) {
    int col = (int)(i & (size_t)hmask);
    float mean = sum[col] * invM;
    float var  = sumsq[col] * invM - mean * mean;
    float v = (Y[i] - mean) * rsqrtf(var + BN_EPS) * gamma[col] + beta[col];
    v = fmaxf(v, 0.0f);
    if (outF) outF[i] = v;
    outH[i] = (h16)v;
  }
}

__device__ __forceinline__ float block_reduce256(float v) {
  __shared__ float red[256];
  red[threadIdx.x] = v;
  __syncthreads();
  for (int s = 128; s > 0; s >>= 1) {
    if ((int)threadIdx.x < s) red[threadIdx.x] += red[threadIdx.x + s];
    __syncthreads();
  }
  return red[0];
}

// sq[row] = sum_c h[row,c]^2 (fp32), one block per row
__global__ __launch_bounds__(256) void row_sumsq_k(const float* __restrict__ Hm,
                                                   float* __restrict__ sq, int D) {
  const int row = blockIdx.x;
  float s = 0.0f;
  for (int c = threadIdx.x; c < D; c += 256) {
    float x = Hm[(size_t)row * D + c];
    s += x * x;
  }
  float t = block_reduce256(s);
  if (threadIdx.x == 0) sq[row] = t;
}

// out[row] = 1 / (K[row,:] . x)   (K symmetric -> serves K and K^T)
__global__ __launch_bounds__(256) void matvec_recip_k(const float* __restrict__ Kmat,
                                                      const float* __restrict__ x,
                                                      float* __restrict__ out, int n) {
  const int row = blockIdx.x;
  const float* kr = Kmat + (size_t)row * n;
  float s = 0.0f;
  for (int j = threadIdx.x; j < n; j += 256) s += kr[j] * x[j];
  float t = block_reduce256(s);
  if (threadIdx.x == 0) out[row] = 1.0f / t;
}

__global__ void fill2_k(float* a, float* b, int n, float val) {
  int i = blockIdx.x * blockDim.x + threadIdx.x;
  if (i < n) { a[i] = val; b[i] = val; }
}

// P16[i,j] = f16( u[i] * K[i,j] * v[j] );  grid = (n/256, n)
__global__ void scale_K_f16_k(const float* __restrict__ Kmat, const float* __restrict__ u,
                              const float* __restrict__ v, h16* __restrict__ P, int n) {
  const int j = blockIdx.x * blockDim.x + threadIdx.x;
  const int i = blockIdx.y;
  P[(size_t)i * n + j] = (h16)(u[i] * Kmat[(size_t)i * n + j] * v[j]);
}

// out[N x M] f16 = transpose(in[M x N] f32). blockDim(32,8). M,N % 32 == 0.
__global__ void transpose_f16_k(const float* __restrict__ in, h16* __restrict__ out,
                                int M, int N) {
  __shared__ float tile[32][33];
  const int c0 = blockIdx.x * 32, r0 = blockIdx.y * 32;
  for (int i = threadIdx.y; i < 32; i += 8)
    tile[i][threadIdx.x] = in[(size_t)(r0 + i) * N + c0 + threadIdx.x];
  __syncthreads();
  for (int i = threadIdx.y; i < 32; i += 8)
    out[(size_t)(c0 + i) * M + r0 + threadIdx.x] = (h16)tile[threadIdx.x][i];
}

// ---------------------------------------------------------------
// Host orchestration
// ---------------------------------------------------------------
static inline size_t alignUp(size_t x) { return (x + 255) & ~(size_t)255; }

extern "C" void kernel_launch(void* const* d_in, const int* in_sizes, int n_in,
                              void* d_out, int out_size, void* d_ws, size_t ws_size,
                              hipStream_t stream) {
  (void)in_sizes; (void)n_in; (void)out_size; (void)ws_size;
  const float* x   = (const float*)d_in[0];
  const float* w1  = (const float*)d_in[1];  const float* b1  = (const float*)d_in[2];
  const float* g1  = (const float*)d_in[3];  const float* be1 = (const float*)d_in[4];
  const float* w2  = (const float*)d_in[5];  const float* b2  = (const float*)d_in[6];
  const float* g2  = (const float*)d_in[7];  const float* be2 = (const float*)d_in[8];
  const float* w3  = (const float*)d_in[9];  const float* b3  = (const float*)d_in[10];
  const float* g3  = (const float*)d_in[11]; const float* be3 = (const float*)d_in[12];
  const float* w4  = (const float*)d_in[13]; const float* b4  = (const float*)d_in[14];

  const int N = N_BATCH;

  // ---- workspace bump allocator (total ~207 MB) ----
  char* base = (char*)d_ws;
  size_t off = 0;
  auto walloc = [&](size_t bytes) -> void* {
    void* p = base + off;
    off = alignUp(off + bytes);
    return p;
  };
  h16*   xh    = (h16*)  walloc((size_t)N * D_IN * 2);          // 10.5 MB
  h16*   w1h   = (h16*)  walloc((size_t)H1_DIM * D_IN * 2);     //  5.2 MB
  h16*   w2h   = (h16*)  walloc((size_t)H2_DIM * H1_DIM * 2);   //  4.2 MB
  h16*   w3h   = (h16*)  walloc((size_t)H3_DIM * H2_DIM * 2);   //  1.0 MB
  h16*   w4h   = (h16*)  walloc((size_t)D_OUT * H3_DIM * 2);    //  0.5 MB
  float* y1    = (float*)walloc((size_t)N * H1_DIM * 4);        // 33.6 MB
  h16*   h1h   = (h16*)  walloc((size_t)N * H1_DIM * 2);        // 16.8 MB
  float* y2    = (float*)walloc((size_t)N * H2_DIM * 4);        // 16.8 MB (h2 f32 in-place)
  h16*   h2h   = (h16*)  walloc((size_t)N * H2_DIM * 2);        //  8.4 MB
  h16*   h2Th  = (h16*)  walloc((size_t)H2_DIM * N * 2);        //  8.4 MB
  float* Kmat  = (float*)walloc((size_t)N * N * 4);             // 67.1 MB
  h16*   Ph    = (h16*)  walloc((size_t)N * N * 2);             // 33.6 MB
  float* sq    = (float*)walloc((size_t)N * 4);
  float* uvec  = (float*)walloc((size_t)N * 4);
  float* vvec  = (float*)walloc((size_t)N * 4);
  float* psum  = (float*)walloc((size_t)BN_SLABS * H1_DIM * 4);
  float* psq   = (float*)walloc((size_t)BN_SLABS * H1_DIM * 4);
  float* csum  = (float*)walloc((size_t)H1_DIM * 4);
  float* csq   = (float*)walloc((size_t)H1_DIM * 4);
  // dead-buffer reuse (lifetimes checked):
  float* y3in  = y1;            // [N, H2] f32  (y1 dead after L1 BN)
  h16*   y3inh = h1h;           // [N, H2] f16  (h1h dead after L2 GEMM)
  float* y3    = (float*)xh;    // [N, H3] f32  (xh dead after L1 GEMM)
  h16*   h3h   = w1h;           // [N, H3] f16  (w1h dead after L1 GEMM)

  const int CB = 256;
  auto gsgrid = [](size_t n) { size_t b = (n + 255) / 256; return (int)(b > 4096 ? 4096 : b); };

  // ---- f32 -> f16 conversions (inputs + weights) ----
  f32_to_f16_k<<<gsgrid((size_t)N * D_IN), CB, 0, stream>>>(x,  xh,  (size_t)N * D_IN);
  f32_to_f16_k<<<gsgrid((size_t)H1_DIM * D_IN), CB, 0, stream>>>(w1, w1h, (size_t)H1_DIM * D_IN);
  f32_to_f16_k<<<gsgrid((size_t)H2_DIM * H1_DIM), CB, 0, stream>>>(w2, w2h, (size_t)H2_DIM * H1_DIM);
  f32_to_f16_k<<<gsgrid((size_t)H3_DIM * H2_DIM), CB, 0, stream>>>(w3, w3h, (size_t)H3_DIM * H2_DIM);
  f32_to_f16_k<<<gsgrid((size_t)D_OUT * H3_DIM), CB, 0, stream>>>(w4, w4h, (size_t)D_OUT * H3_DIM);

  auto bn_layer = [&](const float* Y, const float* g, const float* be,
                      float* outF, h16* outH, int H) {
    dim3 gp(H / 256, BN_SLABS);
    bn_partial_k<<<gp, CB, 0, stream>>>(Y, psum, psq, N, H);
    bn_reduce_k<<<(H + 255) / 256, CB, 0, stream>>>(psum, psq, csum, csq, H);
    bn_relu_k<<<gsgrid((size_t)N * H), CB, 0, stream>>>(
        Y, csum, csq, g, be, outF, outH, (size_t)N * H, H - 1, 1.0f / N);
  };

  // ---- Layer 1: y1 = x @ w1^T + b1 ; BN ; ReLU -> h1h ----
  gemm_nt_wmma<EPI_BIAS><<<dim3(H1_DIM / 64, N / 128), CB, 0, stream>>>(
      xh, w1h, y1, N, H1_DIM, D_IN, b1, nullptr);
  bn_layer(y1, g1, be1, nullptr, h1h, H1_DIM);

  // ---- Layer 2: y2 = h1 @ w2^T + b2 ; BN ; ReLU -> h2 (f32 in-place) + h2h ----
  gemm_nt_wmma<EPI_BIAS><<<dim3(H2_DIM / 64, N / 128), CB, 0, stream>>>(
      h1h, w2h, y2, N, H2_DIM, H1_DIM, b2, nullptr);
  bn_layer(y2, g2, be2, y2, h2h, H2_DIM);

  // ---- Sinkhorn kernel matrix, exp fused into Gram epilogue ----
  row_sumsq_k<<<N, CB, 0, stream>>>(y2, sq, H2_DIM);
  gemm_nt_wmma<EPI_EXPCOST><<<dim3(N / 64, N / 128), CB, 0, stream>>>(
      h2h, h2h, Kmat, N, N, H2_DIM, nullptr, sq);

  // ---- Sinkhorn iterations (C symmetric => K == K^T) ----
  fill2_k<<<(N + 255) / 256, CB, 0, stream>>>(uvec, vvec, N, 1.0f / N);
  for (int it = 0; it < SK_ITERS; ++it) {
    matvec_recip_k<<<N, CB, 0, stream>>>(Kmat, vvec, uvec, N);  // u = 1/(K v)
    matvec_recip_k<<<N, CB, 0, stream>>>(Kmat, uvec, vvec, N);  // v = 1/(K^T u) = 1/(K u)
  }

  // ---- P = diag(u) K diag(v) (to f16); h = P @ h2 via NT gemm on h2^T ----
  scale_K_f16_k<<<dim3(N / 256, N), CB, 0, stream>>>(Kmat, uvec, vvec, Ph, N);
  transpose_f16_k<<<dim3(H2_DIM / 32, N / 32), dim3(32, 8), 0, stream>>>(y2, h2Th, N, H2_DIM);
  gemm_nt_wmma<EPI_NONE><<<dim3(H2_DIM / 64, N / 128), CB, 0, stream>>>(
      Ph, h2Th, y3in, N, H2_DIM, N, nullptr, nullptr);

  // ---- Layer 3: y3 = (P h2) @ w3^T + b3 ; BN ; ReLU -> h3h ----
  f32_to_f16_k<<<gsgrid((size_t)N * H2_DIM), CB, 0, stream>>>(y3in, y3inh, (size_t)N * H2_DIM);
  gemm_nt_wmma<EPI_BIAS><<<dim3(H3_DIM / 64, N / 128), CB, 0, stream>>>(
      y3inh, w3h, y3, N, H3_DIM, H2_DIM, b3, nullptr);
  bn_layer(y3, g3, be3, nullptr, h3h, H3_DIM);

  // ---- Layer 4: out = h3 @ w4^T + b4 (fp32 directly to d_out) ----
  gemm_nt_wmma<EPI_BIAS><<<dim3(D_OUT / 64, N / 128), CB, 0, stream>>>(
      h3h, w4h, (float*)d_out, N, D_OUT, H3_DIM, b4, nullptr);
}
